// CrossWindowBlock_64982855188890
// MI455X (gfx1250) — compile-verified, matrix operand order checked by reading, not run on hardware
//
#include <hip/hip_runtime.h>
#include <math.h>

typedef _Float16 half_t;
typedef __attribute__((ext_vector_type(16))) half_t v16h;
typedef __attribute__((ext_vector_type(8)))  float  v8f;
typedef __attribute__((ext_vector_type(4)))  int    v4i;

union FragA { v16h v; uint4 q[2]; };

__device__ inline v8f zero8() { v8f z; for (int i = 0; i < 8; ++i) z[i] = 0.f; return z; }

// Shift-window region code (Swin mask): 0 for [0,60), 1 for [60,62), 2 for [62,64)
__device__ inline int regc(int x) { return (x < 60) ? 0 : ((x < 62) ? 1 : 2); }

#define DIMC 384
#define MLPH 1536
#define NWIN 256
#define NHEAD 6

// --------------------------- CDNA5 async-to-LDS ----------------------------
#if defined(__has_builtin)
#if __has_builtin(__builtin_amdgcn_global_load_async_to_lds_b128)
#define HAVE_ASYNC_LDS 1
#endif
#if __has_builtin(__builtin_amdgcn_s_wait_asynccnt)
#define HAVE_WAIT_ASYNC 1
#endif
#endif

// Copy 16 bytes (8 halves) global -> LDS, per lane (ASYNCcnt path).
__device__ inline void cp16(half_t* lds, const half_t* g) {
#if defined(HAVE_ASYNC_LDS)
  __builtin_amdgcn_global_load_async_to_lds_b128(
      (__attribute__((address_space(1))) v4i*)g,
      (__attribute__((address_space(3))) v4i*)lds, 0, 0);
#else
  *(uint4*)lds = *(const uint4*)g;
#endif
}

template <int N>
__device__ inline void wait_async() {
#if defined(HAVE_ASYNC_LDS)
#if defined(HAVE_WAIT_ASYNC)
  __builtin_amdgcn_s_wait_asynccnt(N);
#else
  asm volatile("s_wait_asynccnt %0" ::"i"(N) : "memory");
#endif
#endif
}

// ---------------------------------------------------------------------------
// Weight conversion fp32 -> fp16, transposed: dst[n*K + k] = src[k*N + n].
// ---------------------------------------------------------------------------
__global__ __launch_bounds__(256) void cvt_f16_t(const float* __restrict__ src,
                                                 half_t* __restrict__ dst,
                                                 int Kd, int Nd) {
  int i = blockIdx.x * 256 + threadIdx.x;
  if (i < Kd * Nd) {
    int n = i / Kd, k = i - n * Kd;
    dst[i] = (half_t)src[(size_t)k * Nd + n];
  }
}

// ---------------------------------------------------------------------------
// Wave reductions (wave32)
// ---------------------------------------------------------------------------
__device__ inline float wsum32(float v) {
  for (int o = 16; o; o >>= 1) v += __shfl_xor(v, o);
  return v;
}

union HPack { half_t h[4]; uint2 u; };

// ---------------------------------------------------------------------------
// Fused: 3x LayerNorm + roll(-2,-2) + window partition -> fp16 q / kv buffers
// One wave32 per token; float4 loads, uint2 packed f16 stores.
// ---------------------------------------------------------------------------
__global__ __launch_bounds__(256) void ln_window(
    const float* __restrict__ xm, const float* __restrict__ xp,
    const float* __restrict__ g_q,  const float* __restrict__ b_q,
    const float* __restrict__ g_km, const float* __restrict__ b_km,
    const float* __restrict__ g_kp, const float* __restrict__ b_kp,
    half_t* __restrict__ qbuf, half_t* __restrict__ kvbuf) {
  int wave = threadIdx.x >> 5, lane = threadIdx.x & 31;
  int t = blockIdx.x * 8 + wave;          // token in rolled frame (b, h', w')
  int b = t >> 12, rem = t & 4095;
  int hr = rem >> 6, wr = rem & 63;
  int hs = (hr + 2) & 63, ws2 = (wr + 2) & 63;   // roll(-2,-2) source
  size_t src = (((size_t)b << 12) + (hs << 6) + ws2) * DIMC;

  int wh = hr >> 2, ww = wr >> 2, a = hr & 3, c = wr & 3;
  int win = wh * 16 + ww, r = a * 4 + c;
  size_t qrow  = (((size_t)b * NWIN + win) * 16 + r) * DIMC;
  size_t kvrow = (((size_t)b * NWIN + win) * 32 + r) * DIMC;

  float4 v4[3];
  const float* px = xm + src;
  for (int i = 0; i < 3; ++i) v4[i] = *(const float4*)&px[i * 128 + lane * 4];
  float s = 0.f, s2 = 0.f;
  for (int i = 0; i < 3; ++i) s += v4[i].x + v4[i].y + v4[i].z + v4[i].w;
  float mean = wsum32(s) * (1.f / DIMC);
  for (int i = 0; i < 3; ++i) {
    float d0 = v4[i].x - mean, d1 = v4[i].y - mean, d2 = v4[i].z - mean, d3 = v4[i].w - mean;
    s2 += d0 * d0 + d1 * d1 + d2 * d2 + d3 * d3;
  }
  float rstd = rsqrtf(wsum32(s2) * (1.f / DIMC) + 1e-5f);
  for (int i = 0; i < 3; ++i) {
    int c0 = i * 128 + lane * 4;
    float4 g1 = *(const float4*)&g_q[c0],  bb1 = *(const float4*)&b_q[c0];
    float4 g2 = *(const float4*)&g_km[c0], bb2 = *(const float4*)&b_km[c0];
    float n0 = (v4[i].x - mean) * rstd, n1 = (v4[i].y - mean) * rstd;
    float n2 = (v4[i].z - mean) * rstd, n3 = (v4[i].w - mean) * rstd;
    HPack oq, ok;
    oq.h[0] = (half_t)(n0 * g1.x + bb1.x); oq.h[1] = (half_t)(n1 * g1.y + bb1.y);
    oq.h[2] = (half_t)(n2 * g1.z + bb1.z); oq.h[3] = (half_t)(n3 * g1.w + bb1.w);
    ok.h[0] = (half_t)(n0 * g2.x + bb2.x); ok.h[1] = (half_t)(n1 * g2.y + bb2.y);
    ok.h[2] = (half_t)(n2 * g2.z + bb2.z); ok.h[3] = (half_t)(n3 * g2.w + bb2.w);
    *(uint2*)&qbuf[qrow + c0]   = oq.u;
    *(uint2*)&kvbuf[kvrow + c0] = ok.u;
  }
  const float* pp = xp + src;
  for (int i = 0; i < 3; ++i) v4[i] = *(const float4*)&pp[i * 128 + lane * 4];
  s = 0.f; s2 = 0.f;
  for (int i = 0; i < 3; ++i) s += v4[i].x + v4[i].y + v4[i].z + v4[i].w;
  mean = wsum32(s) * (1.f / DIMC);
  for (int i = 0; i < 3; ++i) {
    float d0 = v4[i].x - mean, d1 = v4[i].y - mean, d2 = v4[i].z - mean, d3 = v4[i].w - mean;
    s2 += d0 * d0 + d1 * d1 + d2 * d2 + d3 * d3;
  }
  rstd = rsqrtf(wsum32(s2) * (1.f / DIMC) + 1e-5f);
  size_t kvrow2 = kvrow + (size_t)16 * DIMC;
  for (int i = 0; i < 3; ++i) {
    int c0 = i * 128 + lane * 4;
    float4 g3 = *(const float4*)&g_kp[c0], bb3 = *(const float4*)&b_kp[c0];
    HPack op;
    op.h[0] = (half_t)((v4[i].x - mean) * rstd * g3.x + bb3.x);
    op.h[1] = (half_t)((v4[i].y - mean) * rstd * g3.y + bb3.y);
    op.h[2] = (half_t)((v4[i].z - mean) * rstd * g3.z + bb3.z);
    op.h[3] = (half_t)((v4[i].w - mean) * rstd * g3.w + bb3.w);
    *(uint2*)&kvbuf[kvrow2 + c0] = op.u;
  }
}

// ---------------------------------------------------------------------------
// Plain LayerNorm (fp32 in -> fp16 out), one wave per row, vectorized.
// ---------------------------------------------------------------------------
__global__ __launch_bounds__(256) void ln_simple(const float* __restrict__ x,
                                                 const float* __restrict__ g,
                                                 const float* __restrict__ bb,
                                                 half_t* __restrict__ out) {
  int wave = threadIdx.x >> 5, lane = threadIdx.x & 31;
  size_t row = (size_t)(blockIdx.x * 8 + wave) * DIMC;
  float4 v4[3];
  for (int i = 0; i < 3; ++i) v4[i] = *(const float4*)&x[row + i * 128 + lane * 4];
  float s = 0.f, s2 = 0.f;
  for (int i = 0; i < 3; ++i) s += v4[i].x + v4[i].y + v4[i].z + v4[i].w;
  float mean = wsum32(s) * (1.f / DIMC);
  for (int i = 0; i < 3; ++i) {
    float d0 = v4[i].x - mean, d1 = v4[i].y - mean, d2 = v4[i].z - mean, d3 = v4[i].w - mean;
    s2 += d0 * d0 + d1 * d1 + d2 * d2 + d3 * d3;
  }
  float rstd = rsqrtf(wsum32(s2) * (1.f / DIMC) + 1e-5f);
  for (int i = 0; i < 3; ++i) {
    int c0 = i * 128 + lane * 4;
    float4 g1 = *(const float4*)&g[c0], b1 = *(const float4*)&bb[c0];
    HPack o;
    o.h[0] = (half_t)((v4[i].x - mean) * rstd * g1.x + b1.x);
    o.h[1] = (half_t)((v4[i].y - mean) * rstd * g1.y + b1.y);
    o.h[2] = (half_t)((v4[i].z - mean) * rstd * g1.z + b1.z);
    o.h[3] = (half_t)((v4[i].w - mean) * rstd * g1.w + b1.w);
    *(uint2*)&out[row + c0] = o.u;
  }
}

// ---------------------------------------------------------------------------
// WMMA GEMM: C[M,N](f32 acc) = A[M,K](f16) @ Bt[N,K](f16, pre-transposed)
// 256 threads = 8 waves; block tile 128x64; K-step 64 (8 WMMA per stage).
// Double-buffered LDS (48 KB) with async-to-LDS staging: next tile's 6 async
// loads are issued before a PARTIAL wait (s_wait_asynccnt 6) so they remain
// in flight underneath the WMMAs (async loads retire in order).
// ---------------------------------------------------------------------------
#define BM 128
#define BN 64
#define BK 64
enum { MODE_F16 = 0, MODE_F16_GELU = 1, MODE_WO = 2, MODE_W2 = 3, MODE_VT = 4 };

__global__ __launch_bounds__(256) void gemm_wmma(
    const half_t* __restrict__ A, const half_t* __restrict__ Bt,
    const float* __restrict__ bias, void* __restrict__ Cout,
    int M, int N, int K, int mode, const float* __restrict__ resid) {
  __shared__ __align__(16) half_t Ash[2][BM * BK];   // 2 x 16 KB
  __shared__ __align__(16) half_t Bsh[2][BN * BK];   // 2 x 8 KB, [n][k]
  int tid = threadIdx.x;
  int wave = tid >> 5, lane = tid & 31;
  int li = lane & 15, hf = lane >> 4;
  int gm = blockIdx.y * BM;
  int gn = blockIdx.x * BN;
  int cb = hf * 8;  // lane K-offset per ISA 16-bit A-matrix split

  auto issue = [&](int k0, int buf) {
    // A tile 128x64 halves = 1024 x 16B, 4 per thread
    for (int i = 0; i < 4; ++i) {
      int idx = i * 256 + tid;
      int row = idx >> 3, c8 = idx & 7;
      cp16(&Ash[buf][row * BK + c8 * 8], &A[(size_t)(gm + row) * K + k0 + c8 * 8]);
    }
    // B tile 64x64 halves = 512 x 16B, 2 per thread (Bt already [N][K])
    for (int i = 0; i < 2; ++i) {
      int idx = i * 256 + tid;
      int row = idx >> 3, c8 = idx & 7;
      cp16(&Bsh[buf][row * BK + c8 * 8], &Bt[(size_t)(gn + row) * K + k0 + c8 * 8]);
    }
  };

  v8f acc[4];
  for (int i = 0; i < 4; ++i) acc[i] = zero8();

  int nk = K / BK;
  issue(0, 0);
  for (int t = 0; t < nk; ++t) {
    int cur = t & 1;
    if (t + 1 < nk) {
      issue((t + 1) * BK, cur ^ 1);  // 6 more async loads in flight
      wait_async<6>();               // oldest 6 (current tile) have landed
    } else {
      wait_async<0>();
    }
    __syncthreads();

    const half_t* As = Ash[cur];
    const half_t* Bs = Bsh[cur];
    int arow = wave * 16 + li;
    for (int s = 0; s < 2; ++s) {
      FragA af;
      af.q[0] = *(const uint4*)&As[arow * BK + s * 32 + cb];
      af.q[1] = *(const uint4*)&As[arow * BK + s * 32 + cb + 16];
      FragA bf[4];
      for (int nt = 0; nt < 4; ++nt) {
        int bn = nt * 16 + li;
        bf[nt].q[0] = *(const uint4*)&Bs[bn * BK + s * 32 + cb];
        bf[nt].q[1] = *(const uint4*)&Bs[bn * BK + s * 32 + cb + 16];
      }
      for (int nt = 0; nt < 4; ++nt)
        acc[nt] = __builtin_amdgcn_wmma_f32_16x16x32_f16(
            false, af.v, false, bf[nt].v, (short)0, acc[nt], false, false);
    }
    __syncthreads();  // all waves done reading 'cur' before it is refilled
  }

  // Epilogue. C layout: vgpr r -> row r (lanes<16) / r+8 (lanes>=16); col = lane%16
  for (int nt = 0; nt < 4; ++nt) {
    int col = gn + nt * 16 + li;
    float bvv = bias ? bias[col] : 0.f;
    for (int r = 0; r < 8; ++r) {
      int mrow = gm + wave * 16 + r + hf * 8;
      float val = acc[nt][r] + bvv;
      if (mode == MODE_F16) {
        ((half_t*)Cout)[(size_t)mrow * N + col] = (half_t)val;
      } else if (mode == MODE_F16_GELU) {
        float ge = 0.5f * val * (1.f + erff(val * 0.70710678118f));
        ((half_t*)Cout)[(size_t)mrow * N + col] = (half_t)ge;
      } else if (mode == MODE_WO) {
        // window-order row -> natural row with roll(+2,+2), + residual
        int b = mrow >> 12, rem = mrow & 4095;
        int win = rem >> 4, rr = rem & 15;
        int wh = win >> 4, ww = win & 15;
        int hn = ((wh << 2) + (rr >> 2) + 2) & 63;
        int wn = ((ww << 2) + (rr & 3) + 2) & 63;
        size_t nat = ((size_t)b << 12) + (hn << 6) + wn;
        ((float*)Cout)[nat * DIMC + col] = resid[nat * DIMC + col] + val;
      } else if (mode == MODE_VT) {
        // scatter V into per-(window,head) transposed [n][k] tiles
        int hd = col >> 6, n = col & 63;
        int wi = mrow >> 5, kk = mrow & 31;
        ((half_t*)Cout)[((((size_t)wi * NHEAD + hd) << 6) + n) * 32 + kk] = (half_t)val;
      } else {  // MODE_W2: accumulate into fp32 output (holds x already)
        float* o = (float*)Cout;
        size_t idx = (size_t)mrow * N + col;
        o[idx] = o[idx] + val;
      }
    }
  }
}

// ---------------------------------------------------------------------------
// Attention: one wave per (window, head); 8 waves/block.
// QK^T (4 WMMA) + analytic shift mask + softmax, then P@V (4 WMMA).
// V arrives pre-transposed per (window,head) -> contiguous b128 loads.
// ---------------------------------------------------------------------------
__global__ __launch_bounds__(256) void attention(
    const half_t* __restrict__ Qh, const half_t* __restrict__ Kh,
    const half_t* __restrict__ Vt, half_t* __restrict__ yh) {
  __shared__ __align__(16) half_t Pl[8][16 * 32];
  int wv = threadIdx.x >> 5, lane = threadIdx.x & 31;
  int li = lane & 15, hf = lane >> 4;
  int cb = hf * 8;
  int gw = blockIdx.x * 8 + wv;
  int wi = gw / NHEAD, hd = gw % NHEAD;

  // Q fragments, 2 K-steps of 32 over dh=64
  const half_t* Qbase = Qh + (size_t)wi * 16 * DIMC + hd * 64;
  FragA qf[2];
  for (int s = 0; s < 2; ++s) {
    const half_t* p = Qbase + (size_t)li * DIMC + s * 32 + cb;
    qf[s].q[0] = *(const uint4*)p;
    qf[s].q[1] = *(const uint4*)(p + 16);
  }

  // S = Q K^T : B-fragment column n == K row n (contiguous halves)
  v8f sc[2]; sc[0] = zero8(); sc[1] = zero8();
  const half_t* Kbase = Kh + (size_t)wi * 32 * DIMC + hd * 64;
  for (int nt = 0; nt < 2; ++nt)
    for (int s = 0; s < 2; ++s) {
      FragA kf;
      const half_t* p = Kbase + (size_t)(nt * 16 + li) * DIMC + s * 32 + cb;
      kf.q[0] = *(const uint4*)p;
      kf.q[1] = *(const uint4*)(p + 16);
      sc[nt] = __builtin_amdgcn_wmma_f32_16x16x32_f16(
          false, qf[s].v, false, kf.v, (short)0, sc[nt], false, false);
    }

  // Shift mask + softmax over 32 (mask duplicates across km/kp halves)
  int win = wi & (NWIN - 1);
  int wh = win >> 4, ww = win & 15;
  int hc = wh * 4 + (li >> 2), wc = ww * 4 + (li & 3);
  int codeC = regc(hc) * 3 + regc(wc);
  float p0[8], p1[8];
  for (int r = 0; r < 8; ++r) {
    int m = r + hf * 8;
    int hR = wh * 4 + (m >> 2), wR = ww * 4 + (m & 3);
    float msk = ((regc(hR) * 3 + regc(wR)) == codeC) ? 0.f : -100.f;
    float e0 = sc[0][r] * 0.125f + msk;   // SCALE = 64^-0.5
    float e1 = sc[1][r] * 0.125f + msk;
    float mx = fmaxf(e0, e1);
    for (int o = 8; o; o >>= 1) mx = fmaxf(mx, __shfl_xor(mx, o));
    e0 = __expf(e0 - mx); e1 = __expf(e1 - mx);
    float sm = e0 + e1;
    for (int o = 8; o; o >>= 1) sm += __shfl_xor(sm, o);
    float inv = 1.f / sm;
    p0[r] = e0 * inv; p1[r] = e1 * inv;
  }

  // Redistribute P (C-layout -> A-fragment layout) via wave-private LDS.
  for (int r = 0; r < 8; ++r) {
    int m = r + hf * 8;
    Pl[wv][m * 32 + li]      = (half_t)p0[r];
    Pl[wv][m * 32 + li + 16] = (half_t)p1[r];
  }
  FragA pf;
  pf.q[0] = *(const uint4*)&Pl[wv][li * 32 + cb];
  pf.q[1] = *(const uint4*)&Pl[wv][li * 32 + cb + 16];

  // out = P(16x32) @ V(32x64): 4 N-tiles, single K-step; V pre-transposed.
  const half_t* vb = Vt + (((size_t)wi * NHEAD + hd) << 6) * 32;
  v8f ov[4];
  for (int nt = 0; nt < 4; ++nt) {
    FragA vf;
    vf.q[0] = *(const uint4*)&vb[(nt * 16 + li) * 32 + cb];
    vf.q[1] = *(const uint4*)&vb[(nt * 16 + li) * 32 + cb + 16];
    ov[nt] = __builtin_amdgcn_wmma_f32_16x16x32_f16(
        false, pf.v, false, vf.v, (short)0, zero8(), false, false);
  }

  half_t* yb = yh + (size_t)wi * 16 * DIMC + hd * 64;
  for (int nt = 0; nt < 4; ++nt)
    for (int r = 0; r < 8; ++r) {
      int m = r + hf * 8;
      yb[(size_t)m * DIMC + nt * 16 + li] = (half_t)ov[nt][r];
    }
}

// ---------------------------------------------------------------------------
extern "C" void kernel_launch(void* const* d_in, const int* in_sizes, int n_in,
                              void* d_out, int out_size, void* d_ws, size_t ws_size,
                              hipStream_t stream) {
  const float* x_main = (const float*)d_in[0];
  const float* x_mpmt = (const float*)d_in[1];
  const float* g_q  = (const float*)d_in[2];  const float* b_q  = (const float*)d_in[3];
  const float* g_km = (const float*)d_in[4];  const float* b_km = (const float*)d_in[5];
  const float* g_kp = (const float*)d_in[6];  const float* b_kp = (const float*)d_in[7];
  const float* Wq = (const float*)d_in[8];    const float* bq = (const float*)d_in[9];
  const float* Wk = (const float*)d_in[10];   const float* bk = (const float*)d_in[11];
  const float* Wv = (const float*)d_in[12];   const float* bv = (const float*)d_in[13];
  const float* Wo = (const float*)d_in[14];   const float* bo = (const float*)d_in[15];
  const float* g_mlp = (const float*)d_in[16]; const float* b_mlp = (const float*)d_in[17];
  const float* W1 = (const float*)d_in[18];   const float* b1 = (const float*)d_in[19];
  const float* W2 = (const float*)d_in[20];   const float* b2 = (const float*)d_in[21];

  const size_t TQ = 65536, TKV = 131072;
  char* ws = (char*)d_ws;
  size_t off = 0;
  auto alloc = [&](size_t halves) -> half_t* {
    half_t* p = (half_t*)(ws + off);
    off += ((halves * 2 + 255) / 256) * 256;
    return p;
  };
  half_t* qbuf  = alloc(TQ * DIMC);
  half_t* kvbuf = alloc(TKV * DIMC);
  half_t* Qb    = alloc(TQ * DIMC);
  half_t* Kb    = alloc(TKV * DIMC);
  half_t* Vb    = alloc(TKV * DIMC);   // holds V in transposed (wi,hd,n,k) layout
  half_t* wqt = alloc(DIMC * DIMC);
  half_t* wkt = alloc(DIMC * DIMC);
  half_t* wvt = alloc(DIMC * DIMC);
  half_t* wot = alloc(DIMC * DIMC);
  half_t* w1t = alloc(DIMC * MLPH);
  half_t* w2t = alloc(MLPH * DIMC);
  half_t* yh  = qbuf;   // reuse: qbuf dead after Q GEMM
  half_t* xmh = kvbuf;  // reuse: kvbuf dead after K/V GEMMs
  half_t* hh  = Kb;     // reuse: Kb+Vb (contiguous) == 65536*1536 halves exactly
  float* xout = (float*)d_out;

  // Weight conversions (transposed: Bt[n][k])
  cvt_f16_t<<<(DIMC * DIMC + 255) / 256, 256, 0, stream>>>(Wq, wqt, DIMC, DIMC);
  cvt_f16_t<<<(DIMC * DIMC + 255) / 256, 256, 0, stream>>>(Wk, wkt, DIMC, DIMC);
  cvt_f16_t<<<(DIMC * DIMC + 255) / 256, 256, 0, stream>>>(Wv, wvt, DIMC, DIMC);
  cvt_f16_t<<<(DIMC * DIMC + 255) / 256, 256, 0, stream>>>(Wo, wot, DIMC, DIMC);
  cvt_f16_t<<<(DIMC * MLPH + 255) / 256, 256, 0, stream>>>(W1, w1t, DIMC, MLPH);
  cvt_f16_t<<<(MLPH * DIMC + 255) / 256, 256, 0, stream>>>(W2, w2t, MLPH, DIMC);

  // LN x3 + roll + window
  ln_window<<<8192, 256, 0, stream>>>(x_main, x_mpmt, g_q, b_q, g_km, b_km,
                                      g_kp, b_kp, qbuf, kvbuf);

  // QKV projections
  gemm_wmma<<<dim3(DIMC / BN, TQ / BM), 256, 0, stream>>>(
      qbuf, wqt, bq, Qb, (int)TQ, DIMC, DIMC, MODE_F16, nullptr);
  gemm_wmma<<<dim3(DIMC / BN, TKV / BM), 256, 0, stream>>>(
      kvbuf, wkt, bk, Kb, (int)TKV, DIMC, DIMC, MODE_F16, nullptr);
  gemm_wmma<<<dim3(DIMC / BN, TKV / BM), 256, 0, stream>>>(
      kvbuf, wvt, bv, Vb, (int)TKV, DIMC, DIMC, MODE_VT, nullptr);

  // Windowed cross attention: 16*256*6 = 24576 waves, 8 per block
  attention<<<3072, 256, 0, stream>>>(Qb, Kb, Vb, yh);

  // Output projection + un-roll permutation + residual -> x (fp32 in d_out)
  gemm_wmma<<<dim3(DIMC / BN, TQ / BM), 256, 0, stream>>>(
      yh, wot, bo, xout, (int)TQ, DIMC, DIMC, MODE_WO, x_main);

  // MLP
  ln_simple<<<8192, 256, 0, stream>>>(xout, g_mlp, b_mlp, xmh);
  gemm_wmma<<<dim3(MLPH / BN, TQ / BM), 256, 0, stream>>>(
      xmh, w1t, b1, hh, (int)TQ, MLPH, DIMC, MODE_F16_GELU, nullptr);
  gemm_wmma<<<dim3(DIMC / BN, TQ / BM), 256, 0, stream>>>(
      hh, w2t, b2, xout, (int)TQ, DIMC, MLPH, MODE_W2, nullptr);
}
